// lstm_latent_3_68307159876124
// MI455X (gfx1250) — matrix operand order repaired and checked by reading, not compiled, not run on hardware
//
#include <hip/hip_runtime.h>
#include <cstddef>

// ---------------- problem sizes ----------------
static constexpr int B_   = 128;
static constexpr int T_   = 256;
static constexpr int DLAT = 256;
static constexpr int HS   = 512;   // lstm1/lstm2 hidden
static constexpr int HB   = 1024;  // lstm3 hidden

static constexpr int NBLK = 32;
static constexpr int TPB  = 256;   // 8 waves per block (wave32)

// LDS staging: K-chunk width and padded row strides (bf16 elements)
static constexpr int KC  = 64;
static constexpr int SAP = KC + 8;   // A row stride (pad 16B to spread banks)
static constexpr int SBP = KC + 8;   // B row stride

// ---------------- WMMA types ----------------
typedef __attribute__((ext_vector_type(16))) __bf16       v16bf;
typedef __attribute__((ext_vector_type(8)))  float        v8f;
typedef __attribute__((ext_vector_type(4)))  unsigned int v4u;

// ---------------- workspace layout (bytes) ----------------
static constexpr size_t OFF_W1I = 0;                         // bf16 [4HS][DLAT]
static constexpr size_t SZ_W1I  = (size_t)4*HS*DLAT*2;
static constexpr size_t OFF_W1H = OFF_W1I + SZ_W1I;          // bf16 [4HS][HS]
static constexpr size_t SZ_W1H  = (size_t)4*HS*HS*2;
static constexpr size_t OFF_W2I = OFF_W1H + SZ_W1H;          // bf16 [4HS][HS]
static constexpr size_t SZ_W2I  = (size_t)4*HS*HS*2;
static constexpr size_t OFF_W2H = OFF_W2I + SZ_W2I;          // bf16 [4HS][HS]
static constexpr size_t SZ_W2H  = (size_t)4*HS*HS*2;
static constexpr size_t OFF_W3I = OFF_W2H + SZ_W2H;          // bf16 [4HB][HS]
static constexpr size_t SZ_W3I  = (size_t)4*HB*HS*2;
static constexpr size_t OFF_W3H = OFF_W3I + SZ_W3I;          // bf16 [4HB][HB]
static constexpr size_t SZ_W3H  = (size_t)4*HB*HB*2;
static constexpr size_t OFF_WL  = OFF_W3H + SZ_W3H;          // bf16 [DLAT][HB]
static constexpr size_t SZ_WL   = (size_t)DLAT*HB*2;
// combined biases (fp32)
static constexpr size_t OFF_B1  = OFF_WL + SZ_WL;
static constexpr size_t OFF_B2  = OFF_B1 + (size_t)4*HS*4;
static constexpr size_t OFF_B3  = OFF_B2 + (size_t)4*HS*4;
static constexpr size_t OFF_BL  = OFF_B3 + (size_t)4*HB*4;
// zero-initialized state (memset each launch)
static constexpr size_t OFF_STATE = OFF_BL + (size_t)DLAT*4;
static constexpr size_t OFF_H1  = OFF_STATE;                 // bf16 2*[B][HS]
static constexpr size_t SZ_H1   = (size_t)2*B_*HS*2;
static constexpr size_t OFF_H2  = OFF_H1 + SZ_H1;            // bf16 2*[B][HS]
static constexpr size_t SZ_H2   = (size_t)2*B_*HS*2;
static constexpr size_t OFF_H3  = OFF_H2 + SZ_H2;            // bf16 2*[B][HB]
static constexpr size_t SZ_H3   = (size_t)2*B_*HB*2;
static constexpr size_t OFF_C1  = OFF_H3 + SZ_H3;            // f32 [B][HS]
static constexpr size_t SZ_C1   = (size_t)B_*HS*4;
static constexpr size_t OFF_C2  = OFF_C1 + SZ_C1;            // f32 [B][HS]
static constexpr size_t SZ_C2   = (size_t)B_*HS*4;
static constexpr size_t OFF_C3  = OFF_C2 + SZ_C2;            // f32 [B][HB]
static constexpr size_t SZ_C3   = (size_t)B_*HB*4;
static constexpr size_t OFF_BAR = OFF_C3 + SZ_C3;            // barrier cnt/flag
static constexpr size_t SZ_BAR  = 256;
static constexpr size_t STATE_SIZE = OFF_BAR + SZ_BAR - OFF_STATE;

// ---------------- small helpers ----------------
__device__ __forceinline__ unsigned short f2bf(float f) {
    unsigned u = __float_as_uint(f);
    unsigned r = u + 0x7FFFu + ((u >> 16) & 1u);   // round-to-nearest-even
    return (unsigned short)(r >> 16);
}
__device__ __forceinline__ float sigm(float x) {
    return 1.0f / (1.0f + __expf(-x));
}
__device__ __forceinline__ float tanh_fast(float x) {
    float a = __builtin_fabsf(x);
    float e = __expf(2.0f * a);
    float t = 1.0f - 2.0f / (e + 1.0f);
    return __builtin_copysignf(t, x);
}
__device__ __forceinline__ v8f wmma_bf16(v16bf a, v16bf b, v8f c) {
    return __builtin_amdgcn_wmma_f32_16x16x32_bf16(
        false, a, false, b, (short)0, c, false, false);
}

// ---------------- LDS fragment loaders (16-bit WMMA operand layouts) ----------------
// A 16x32: lane<16 -> K = kk+0..7 / kk+16..23 ; lane>=16 -> kk+8..15 / kk+24..31
__device__ __forceinline__ v16bf lds_a(const unsigned short* buf, int m0, int kk, int lane) {
    int l = lane & 15, hi = lane >> 4;
    const unsigned short* p = buf + (m0 + l) * SAP + kk + hi * 8;
    union { v16bf v; v4u q[2]; } u;
    u.q[0] = *(const v4u*)p;
    u.q[1] = *(const v4u*)(p + 16);
    return u.v;
}
// B 32x16 (B[k][n] = W[n][k]): lane<16 -> K=kk+0..7 / kk+16..23, lane>=16 -> K=kk+8..15 / kk+24..31; N=lane&15
__device__ __forceinline__ v16bf lds_b(const unsigned short* buf, int g, int kk, int lane) {
    int n = lane & 15, kh = (lane >> 4) * 8;
    const unsigned short* p = buf + (g * 16 + n) * SBP + kk + kh;
    union { v16bf v; v4u q[2]; } u;
    u.q[0] = *(const v4u*)p;
    u.q[1] = *(const v4u*)(p + 16);
    return u.v;
}

// ---------------- grid-wide barrier (sense reversing; state zeroed per launch) ----------
__device__ __forceinline__ void grid_barrier(unsigned int* cnt, unsigned int* flag,
                                             unsigned int sense) {
    __threadfence();
    __syncthreads();
    if (threadIdx.x == 0) {
        unsigned nblk = gridDim.x;
        unsigned old = __hip_atomic_fetch_add(cnt, 1u, __ATOMIC_ACQ_REL,
                                              __HIP_MEMORY_SCOPE_AGENT);
        if (old == nblk - 1u) {
            __hip_atomic_store(cnt, 0u, __ATOMIC_RELAXED, __HIP_MEMORY_SCOPE_AGENT);
            __hip_atomic_store(flag, sense, __ATOMIC_RELEASE, __HIP_MEMORY_SCOPE_AGENT);
        } else {
            while (__hip_atomic_load(flag, __ATOMIC_ACQUIRE,
                                     __HIP_MEMORY_SCOPE_AGENT) != sense) {
                __builtin_amdgcn_s_sleep(1);
            }
        }
    }
    __syncthreads();
    __threadfence();
}

// ---------------- double-buffered LDS K-panel ------------------------------------------
// acc[g] += A(tile m0) x B(row group g at n0), k in [0,K).  Whole block cooperatively
// stages A[128 x KC] and B[GATES*16 x KC] chunks into LDS; global loads for chunk c+1
// are issued before computing chunk c; one __syncthreads per chunk (writes always go to
// the opposite buffer, and all chunk counts are even, so panel boundaries are race-free).
template <bool XF32, int GATES>
__device__ __forceinline__ void panel_run(
    v8f (&acc)[GATES],
    unsigned short* sA, unsigned short* sB,   // [2][128*SAP], [2][64*SBP]
    const float* __restrict__ Af, const unsigned short* __restrict__ Ab, int lda,
    const unsigned short* __restrict__ W, int ldw, int H, int n0,
    int K, int m0, int lane, int tid)
{
    // staging assignments: A -> 2 threads/row (32 elems each); B -> 4 threads/row (16 each)
    const int arow = tid >> 1, acol = (tid & 1) * 32;
    const int brow = tid >> 2, bcol = (tid & 3) * 16;
    const int bg = brow >> 4, brr = brow & 15;
    const bool bact = (GATES == 4) || (tid < 64);   // wave-uniform B-staging guard
    const unsigned short* wrow = W + (size_t)(bg * H + n0 + brr) * ldw;

    v4u ar[8];   // raw A piece (4 used for bf16, 8 for f32)
    v4u brg[2];  // raw B piece

    // prologue: global loads for chunk 0
    {
        if (bact) {
            const unsigned short* wp = wrow + bcol;
            brg[0] = *(const v4u*)wp; brg[1] = *(const v4u*)(wp + 8);
        }
        if constexpr (XF32) {
            const float* ap = Af + (size_t)arow * lda + acol;
#pragma unroll
            for (int j = 0; j < 8; ++j) ar[j] = *(const v4u*)(ap + 4 * j);
        } else {
            const unsigned short* ap = Ab + (size_t)arow * lda + acol;
#pragma unroll
            for (int j = 0; j < 4; ++j) ar[j] = *(const v4u*)(ap + 8 * j);
        }
    }

    const int nch = K / KC;   // always even for our K values
#pragma unroll 1
    for (int c = 0; c < nch; ++c) {
        unsigned short* bufA = sA + (c & 1) * (128 * SAP);
        unsigned short* bufB = sB + (c & 1) * (64 * SBP);

        // commit staged regs to LDS
        if (bact) {
            *(v4u*)(bufB + brow * SBP + bcol)     = brg[0];
            *(v4u*)(bufB + brow * SBP + bcol + 8) = brg[1];
        }
        if constexpr (XF32) {
            union { v4u q[8]; float f[32]; } rw;
#pragma unroll
            for (int j = 0; j < 8; ++j) rw.q[j] = ar[j];
            union { v4u q[4]; unsigned short s[32]; } cv;
#pragma unroll
            for (int j = 0; j < 32; ++j) cv.s[j] = f2bf(rw.f[j]);
#pragma unroll
            for (int j = 0; j < 4; ++j)
                *(v4u*)(bufA + arow * SAP + acol + 8 * j) = cv.q[j];
        } else {
#pragma unroll
            for (int j = 0; j < 4; ++j)
                *(v4u*)(bufA + arow * SAP + acol + 8 * j) = ar[j];
        }

        // issue global loads for chunk c+1 (hidden behind this chunk's compute)
        if (c + 1 < nch) {
            const int cb = (c + 1) * KC;
            if (bact) {
                const unsigned short* wp = wrow + cb + bcol;
                brg[0] = *(const v4u*)wp; brg[1] = *(const v4u*)(wp + 8);
            }
            if constexpr (XF32) {
                const float* ap = Af + (size_t)arow * lda + cb + acol;
#pragma unroll
                for (int j = 0; j < 8; ++j) ar[j] = *(const v4u*)(ap + 4 * j);
            } else {
                const unsigned short* ap = Ab + (size_t)arow * lda + cb + acol;
#pragma unroll
                for (int j = 0; j < 4; ++j) ar[j] = *(const v4u*)(ap + 8 * j);
            }
        }

        __syncthreads();   // chunk c fully staged

        // compute KC/32 = 2 k-steps from LDS
#pragma unroll
        for (int kk = 0; kk < KC; kk += 32) {
            v16bf a = lds_a(bufA, m0, kk, lane);
#pragma unroll
            for (int g = 0; g < GATES; ++g) {
                v16bf b = lds_b(bufB, g, kk, lane);
                acc[g] = wmma_bf16(a, b, acc[g]);
            }
        }
    }
}

// ---------------- one LSTM layer for one timestep (block-cooperative) ----------------
template <int ITER, bool XF32>
__device__ __forceinline__ void lstm_layer_lds(
    int blk, int wv, int tid, int lane,
    int H, int K1,
    const float* __restrict__ Af, const unsigned short* __restrict__ Ab, int lda1,
    const unsigned short* __restrict__ A2,   // recurrent h (prev step), [B][H]
    const unsigned short* __restrict__ W1,   // [4H][K1]
    const unsigned short* __restrict__ W2,   // [4H][H]
    const float* __restrict__ bias,          // [4H]
    float* __restrict__ c,                   // [B][H]
    unsigned short* __restrict__ hout,       // [B][H] next buffer, bf16
    unsigned short* sA, unsigned short* sB)
{
    const int m0 = wv * 16;
#pragma unroll 1
    for (int i = 0; i < ITER; ++i) {
        const int n0 = (blk + NBLK * i) * 16;    // block-uniform output column tile

        v8f acc[4];
#pragma unroll
        for (int g = 0; g < 4; ++g) {
            v8f z = {0.f, 0.f, 0.f, 0.f, 0.f, 0.f, 0.f, 0.f};
            acc[g] = z;
        }

        panel_run<XF32, 4>(acc, sA, sB, Af, Ab, lda1, W1, K1, H, n0, K1, m0, lane, tid);
        panel_run<false, 4>(acc, sA, sB, nullptr, A2, H, W2, H, H, n0, H, m0, lane, tid);

        // elementwise LSTM cell in registers (i,f,g,o tiles share the C/D layout)
        int l = lane & 15, hi = lane >> 4;
        int n = n0 + l;
        float bi = bias[n], bff = bias[H + n], bg = bias[2 * H + n], bo = bias[3 * H + n];
#pragma unroll
        for (int q = 0; q < 8; ++q) {
            int m = m0 + q + hi * 8;
            float gi = acc[0][q] + bi;
            float gf = acc[1][q] + bff;
            float gg = acc[2][q] + bg;
            float go = acc[3][q] + bo;
            size_t idx = (size_t)m * H + n;
            float cn = sigm(gf) * c[idx] + sigm(gi) * tanh_fast(gg);
            c[idx] = cn;
            hout[idx] = f2bf(sigm(go) * tanh_fast(cn));
        }
    }
}

// ---------------- output Linear via the same LDS pipeline ------------------------------
// Blocks 0..15 each own one block-uniform column tile n0 = blk*16 (16 nt x 8 mt = 128).
__device__ __forceinline__ void lin_phase_lds(
    int blk, int wv, int tid, int lane,
    const unsigned short* __restrict__ h3,   // [B][HB] bf16
    const unsigned short* __restrict__ Wl,   // [DLAT][HB] bf16
    const float* __restrict__ bl,            // [DLAT]
    float* __restrict__ out,                 // [B][DLAT]
    unsigned short* sA, unsigned short* sB)
{
    if (blk >= DLAT / 16) return;            // blocks 16..31 idle (block-local syncs only)
    const int n0 = blk * 16, m0 = wv * 16;

    v8f acc[1];
    {
        v8f z = {0.f, 0.f, 0.f, 0.f, 0.f, 0.f, 0.f, 0.f};
        acc[0] = z;
    }
    panel_run<false, 1>(acc, sA, sB, nullptr, h3, HB, Wl, HB, /*H unused*/ 0, n0,
                        HB, m0, lane, tid);

    int l = lane & 15, hi = lane >> 4;
    int n = n0 + l;
    float bb = bl[n];
#pragma unroll
    for (int q = 0; q < 8; ++q) {
        int m = m0 + q + hi * 8;
        out[(size_t)m * DLAT + n] = acc[0][q] + bb;
    }
}

__device__ __forceinline__ void cvt_arr(unsigned short* dst, const float* src,
                                        int n, int gtid, int gsz) {
    for (int i = gtid; i < n; i += gsz) dst[i] = f2bf(src[i]);
}

// ---------------- persistent kernel ----------------
__global__ __launch_bounds__(TPB) void lstm3_persistent(
    const float* __restrict__ X,
    const float* __restrict__ Wih1, const float* __restrict__ Whh1,
    const float* __restrict__ bih1, const float* __restrict__ bhh1,
    const float* __restrict__ Wih2, const float* __restrict__ Whh2,
    const float* __restrict__ bih2, const float* __restrict__ bhh2,
    const float* __restrict__ Wih3, const float* __restrict__ Whh3,
    const float* __restrict__ bih3, const float* __restrict__ bhh3,
    const float* __restrict__ Wlin, const float* __restrict__ blin,
    float* __restrict__ out, unsigned char* __restrict__ ws)
{
    __shared__ __align__(16) unsigned short sA[2 * 128 * SAP];  // 36 KB
    __shared__ __align__(16) unsigned short sB[2 * 64 * SBP];   // 18 KB

    unsigned short* W1I = (unsigned short*)(ws + OFF_W1I);
    unsigned short* W1H = (unsigned short*)(ws + OFF_W1H);
    unsigned short* W2I = (unsigned short*)(ws + OFF_W2I);
    unsigned short* W2H = (unsigned short*)(ws + OFF_W2H);
    unsigned short* W3I = (unsigned short*)(ws + OFF_W3I);
    unsigned short* W3H = (unsigned short*)(ws + OFF_W3H);
    unsigned short* WL  = (unsigned short*)(ws + OFF_WL);
    float* b1 = (float*)(ws + OFF_B1);
    float* b2 = (float*)(ws + OFF_B2);
    float* b3 = (float*)(ws + OFF_B3);
    float* bl = (float*)(ws + OFF_BL);
    unsigned short* h1 = (unsigned short*)(ws + OFF_H1);
    unsigned short* h2 = (unsigned short*)(ws + OFF_H2);
    unsigned short* h3 = (unsigned short*)(ws + OFF_H3);
    float* c1 = (float*)(ws + OFF_C1);
    float* c2 = (float*)(ws + OFF_C2);
    float* c3 = (float*)(ws + OFF_C3);
    unsigned int* cnt = (unsigned int*)(ws + OFF_BAR);
    unsigned int* flg = cnt + 1;

    const int tid    = threadIdx.x;
    const int blk    = blockIdx.x;
    const int gtid   = blk * blockDim.x + tid;
    const int gsz    = gridDim.x * blockDim.x;
    const int lane   = tid & 31;
    const int wv     = tid >> 5;

    // ---- pre-pass: fp32 -> bf16 weights, combined biases ----
    cvt_arr(W1I, Wih1, 4 * HS * DLAT, gtid, gsz);
    cvt_arr(W1H, Whh1, 4 * HS * HS,   gtid, gsz);
    cvt_arr(W2I, Wih2, 4 * HS * HS,   gtid, gsz);
    cvt_arr(W2H, Whh2, 4 * HS * HS,   gtid, gsz);
    cvt_arr(W3I, Wih3, 4 * HB * HS,   gtid, gsz);
    cvt_arr(W3H, Whh3, 4 * HB * HB,   gtid, gsz);
    cvt_arr(WL,  Wlin, DLAT * HB,     gtid, gsz);
    for (int i = gtid; i < 4 * HS; i += gsz) b1[i] = bih1[i] + bhh1[i];
    for (int i = gtid; i < 4 * HS; i += gsz) b2[i] = bih2[i] + bhh2[i];
    for (int i = gtid; i < 4 * HB; i += gsz) b3[i] = bih3[i] + bhh3[i];
    for (int i = gtid; i < DLAT;   i += gsz) bl[i] = blin[i];

    unsigned sense = 0;
    grid_barrier(cnt, flg, sense ^= 1u);

    // ---- time loop: 3 grid barriers per step ----
#pragma unroll 1
    for (int t = 0; t < T_; ++t) {
        const int cur = t & 1, nxt = cur ^ 1;
        const unsigned short* h1c = h1 + (size_t)cur * B_ * HS;
        unsigned short*       h1n = h1 + (size_t)nxt * B_ * HS;
        const unsigned short* h2c = h2 + (size_t)cur * B_ * HS;
        unsigned short*       h2n = h2 + (size_t)nxt * B_ * HS;
        const unsigned short* h3c = h3 + (size_t)cur * B_ * HB;
        unsigned short*       h3n = h3 + (size_t)nxt * B_ * HB;

        // phase 1: Linear for step t-1 (reads h3 cur buffer) + LSTM1 for step t
        if (t > 0)
            lin_phase_lds(blk, wv, tid, lane, h3c, WL, bl,
                          out + (size_t)(t - 1) * B_ * DLAT, sA, sB);
        lstm_layer_lds<1, true>(blk, wv, tid, lane, HS, DLAT,
                                X + (size_t)t * DLAT, nullptr, T_ * DLAT,
                                h1c, W1I, W1H, b1, c1, h1n, sA, sB);
        grid_barrier(cnt, flg, sense ^= 1u);

        // phase 2: LSTM2 (input = new h1)
        lstm_layer_lds<1, false>(blk, wv, tid, lane, HS, HS,
                                 nullptr, h1n, HS,
                                 h2c, W2I, W2H, b2, c2, h2n, sA, sB);
        grid_barrier(cnt, flg, sense ^= 1u);

        // phase 3: LSTM3 (input = new h2); two block-uniform nt passes cover 64 nt tiles
        lstm_layer_lds<2, false>(blk, wv, tid, lane, HB, HS,
                                 nullptr, h2n, HS,
                                 h3c, W3I, W3H, b3, c3, h3n, sA, sB);
        grid_barrier(cnt, flg, sense ^= 1u);
    }

    // final output for t = T-1 (its h3 lives in buffer (T-1)&1 ^ 1 == 0)
    lin_phase_lds(blk, wv, tid, lane, h3 + (size_t)(((T_ - 1) & 1) ^ 1) * B_ * HB,
                  WL, bl, out + (size_t)(T_ - 1) * B_ * DLAT, sA, sB);
}

// ---------------- host entry ----------------
extern "C" void kernel_launch(void* const* d_in, const int* in_sizes, int n_in,
                              void* d_out, int out_size, void* d_ws, size_t ws_size,
                              hipStream_t stream) {
    (void)in_sizes; (void)n_in; (void)out_size; (void)ws_size;
    unsigned char* ws = (unsigned char*)d_ws;

    // zero h/c state and barrier counters every call (graph-capture safe)
    hipMemsetAsync(ws + OFF_STATE, 0, STATE_SIZE, stream);

    lstm3_persistent<<<dim3(NBLK), dim3(TPB), 0, stream>>>(
        (const float*)d_in[0],
        (const float*)d_in[1],  (const float*)d_in[2],
        (const float*)d_in[3],  (const float*)d_in[4],
        (const float*)d_in[5],  (const float*)d_in[6],
        (const float*)d_in[7],  (const float*)d_in[8],
        (const float*)d_in[9],  (const float*)d_in[10],
        (const float*)d_in[11], (const float*)d_in[12],
        (const float*)d_in[13], (const float*)d_in[14],
        (float*)d_out, ws);
}
